// MDIA_70033736729240
// MI455X (gfx1250) — matrix operand order repaired and checked by reading, not compiled, not run on hardware
//
#include <hip/hip_runtime.h>
#include <hip/hip_bf16.h>

typedef __attribute__((ext_vector_type(16))) _Float16 v16h;
typedef __attribute__((ext_vector_type(8)))  float    v8f;

#define D_MODEL 1024
#define NHEAD   16
#define HDIM    64
#define SEQ     2048
#define BATCH   4
#define ROWS    (BATCH * SEQ)        // 8192
#define SCALE   0.03125f             // 1024^-0.5
#define PSTRIDE 40                   // halfs per LDS row (16B-aligned, conflict-free)

union F16x16 { v16h v; uint4 q[2]; };

// ---- fragment loaders (CDNA5 16-bit WMMA layouts, ISA 7.12.2) ----
// A 16x32: lane row = L%16; chunk0 halfs K=[8*hi..8*hi+7], chunk1 K=[16+8*hi..+7]
__device__ inline v16h load_a_frag(const _Float16* __restrict__ base, int row_stride,
                                   int row, int k0, int hi) {
    F16x16 u;
    const _Float16* p = base + (size_t)row * row_stride + k0 + 8 * hi;
    u.q[0] = *(const uint4*)p;
    u.q[1] = *(const uint4*)(p + 16);
    return u.v;
}
// B 32x16: lane col = L%16; chunk0 halfs K=[8*hi..8*hi+7], chunk1 K=[16+8*hi..+7]
__device__ inline v16h load_b_frag(const _Float16* __restrict__ base, int row_stride,
                                   int col_row, int k0, int hi) {
    F16x16 u;
    const _Float16* p = base + (size_t)col_row * row_stride + k0 + 8 * hi;
    u.q[0] = *(const uint4*)p;
    u.q[1] = *(const uint4*)(p + 16);
    return u.v;
}

__device__ inline v8f wmma_f16(v16h a, v16h b, v8f c) {
    return __builtin_amdgcn_wmma_f32_16x16x32_f16(false, a, false, b, (short)0, c,
                                                  false, false);
}

// ---------------- kernel 0: f32 -> f16 convert ----------------
__global__ void MDIA_cvt_f16(const float* __restrict__ src, _Float16* __restrict__ dst, int n) {
    int i = blockIdx.x * blockDim.x + threadIdx.x;
    if (i < n) dst[i] = (_Float16)src[i];
}

// ---------------- kernel 1: fused QKV projection (32x64 per wave) ----------------
// qkv[row, col] = sum_k xh[row,k] * wqh[col,k]; scatter to Q/K/V^T head buffers.
__global__ __launch_bounds__(256) void MDIA_qkv_gemm(
        const _Float16* __restrict__ xh, const _Float16* __restrict__ wqh,
        _Float16* __restrict__ qb, _Float16* __restrict__ kb, _Float16* __restrict__ vtb) {
    const int lane = threadIdx.x & 31, wv = threadIdx.x >> 5;
    const int hi = lane >> 4, ln = lane & 15;
    const int m0 = blockIdx.x * 32;
    const int n0 = (blockIdx.y * 8 + wv) * 64;

    v8f acc[2][4] = {};
    for (int k0 = 0; k0 < D_MODEL; k0 += 32) {
        v16h a0 = load_a_frag(xh, D_MODEL, m0 + ln, k0, hi);
        v16h a1 = load_a_frag(xh, D_MODEL, m0 + 16 + ln, k0, hi);
#pragma unroll
        for (int nt = 0; nt < 4; ++nt) {
            v16h b = load_b_frag(wqh, D_MODEL, n0 + 16 * nt + ln, k0, hi);
            acc[0][nt] = wmma_f16(a0, b, acc[0][nt]);
            acc[1][nt] = wmma_f16(a1, b, acc[1][nt]);
        }
    }
#pragma unroll
    for (int nt = 0; nt < 4; ++nt) {
        int col = n0 + 16 * nt + ln;
        int three = col >> 10, h = (col >> 6) & 15, hd = col & 63;
#pragma unroll
        for (int mt = 0; mt < 2; ++mt) {
#pragma unroll
            for (int r = 0; r < 8; ++r) {
                int row = m0 + 16 * mt + r + 8 * hi;
                int bidx = row >> 11, s = row & 2047;
                _Float16 v = (_Float16)acc[mt][nt][r];
                size_t bh = (size_t)(bidx * NHEAD + h);
                if (three == 0)      qb[(bh * SEQ + s) * HDIM + hd] = v;
                else if (three == 1) kb[(bh * SEQ + s) * HDIM + hd] = v;
                else                 vtb[(bh * HDIM + hd) * SEQ + s] = v;
            }
        }
    }
}

// ---------------- kernel 2: flash attention, two 16-row q tiles per wave ----------------
__global__ __launch_bounds__(128) void MDIA_attn(
        const _Float16* __restrict__ qb, const _Float16* __restrict__ kb,
        const _Float16* __restrict__ vtb, _Float16* __restrict__ aout) {
    __shared__ _Float16 plds[4 * 2 * 16 * PSTRIDE];   // 4 waves x 2 tiles
    const int lane = threadIdx.x & 31, wv = threadIdx.x >> 5;
    const int hi = lane >> 4, ln = lane & 15;
    const int bh = blockIdx.y, bidx = bh >> 4, h = bh & 15;
    const int q0 = blockIdx.x * 128 + wv * 32;

    const _Float16* qp = qb  + (size_t)bh * SEQ * HDIM;
    const _Float16* kp = kb  + (size_t)bh * SEQ * HDIM;
    const _Float16* vp = vtb + (size_t)bh * HDIM * SEQ;

    // Q fragments: 2 q-tiles x {hd 0-31, hd 32-63}, resident for the whole j-loop
    v16h aq[2][2];
#pragma unroll
    for (int qt = 0; qt < 2; ++qt) {
        aq[qt][0] = load_a_frag(qp, HDIM, q0 + 16 * qt + ln, 0, hi);
        aq[qt][1] = load_a_frag(qp, HDIM, q0 + 16 * qt + ln, 32, hi);
    }

    float mrun[2][8], lrun[2][8];
#pragma unroll
    for (int qt = 0; qt < 2; ++qt)
#pragma unroll
        for (int r = 0; r < 8; ++r) { mrun[qt][r] = -INFINITY; lrun[qt][r] = 0.f; }
    v8f acc[2][4] = {};

    for (int j0 = 0; j0 < SEQ; j0 += 32) {
        // shared K fragments for this 32-wide j tile
        v16h bk[2][2];
#pragma unroll
        for (int nt = 0; nt < 2; ++nt) {
            bk[nt][0] = load_b_frag(kp, HDIM, j0 + 16 * nt + ln, 0, hi);
            bk[nt][1] = load_b_frag(kp, HDIM, j0 + 16 * nt + ln, 32, hi);
        }
#pragma unroll
        for (int qt = 0; qt < 2; ++qt) {
            _Float16* pl = plds + (wv * 2 + qt) * 16 * PSTRIDE;
            v8f sc[2];
#pragma unroll
            for (int nt = 0; nt < 2; ++nt) {
                v8f c = {};
                c = wmma_f16(aq[qt][0], bk[nt][0], c);
                c = wmma_f16(aq[qt][1], bk[nt][1], c);
                sc[nt] = c;
            }
            // online softmax; write P straight into the per-tile LDS scratch
#pragma unroll
            for (int r = 0; r < 8; ++r) {
                float s0 = sc[0][r] * SCALE, s1 = sc[1][r] * SCALE;
                float mx = fmaxf(s0, s1);
                mx = fmaxf(mx, __shfl_xor(mx, 1, 32));
                mx = fmaxf(mx, __shfl_xor(mx, 2, 32));
                mx = fmaxf(mx, __shfl_xor(mx, 4, 32));
                mx = fmaxf(mx, __shfl_xor(mx, 8, 32));
                float nm = fmaxf(mrun[qt][r], mx);
                float alpha = __expf(mrun[qt][r] - nm);
                mrun[qt][r] = nm;
                float p0 = __expf(s0 - nm), p1 = __expf(s1 - nm);
                float rs = p0 + p1;
                rs += __shfl_xor(rs, 1, 32);
                rs += __shfl_xor(rs, 2, 32);
                rs += __shfl_xor(rs, 4, 32);
                rs += __shfl_xor(rs, 8, 32);
                lrun[qt][r] = lrun[qt][r] * alpha + rs;
#pragma unroll
                for (int t = 0; t < 4; ++t) acc[qt][t][r] *= alpha;
                pl[(r + 8 * hi) * PSTRIDE + ln]      = (_Float16)p0;
                pl[(r + 8 * hi) * PSTRIDE + 16 + ln] = (_Float16)p1;
            }
        }
        asm volatile("s_wait_dscnt 0x0" ::: "memory");
        // reshape P tiles into A fragments
        v16h pa[2];
#pragma unroll
        for (int qt = 0; qt < 2; ++qt) {
            F16x16 ua;
            const _Float16* pr = plds + (wv * 2 + qt) * 16 * PSTRIDE + ln * PSTRIDE + 8 * hi;
            ua.q[0] = *(const uint4*)pr;
            ua.q[1] = *(const uint4*)(pr + 16);
            pa[qt] = ua.v;
        }
        // P @ V : V^T rows are contiguous in j; V fragments shared across q-tiles
#pragma unroll
        for (int t = 0; t < 4; ++t) {
            v16h bv = load_b_frag(vp, SEQ, 16 * t + ln, j0, hi);
            acc[0][t] = wmma_f16(pa[0], bv, acc[0][t]);
            acc[1][t] = wmma_f16(pa[1], bv, acc[1][t]);
        }
    }
    // normalize and write attention output [B*S][D] as f16
#pragma unroll
    for (int qt = 0; qt < 2; ++qt) {
#pragma unroll
        for (int r = 0; r < 8; ++r) {
            float inv = 1.0f / lrun[qt][r];
            int row = bidx * SEQ + q0 + 16 * qt + r + 8 * hi;
#pragma unroll
            for (int t = 0; t < 4; ++t) {
                int col = h * HDIM + 16 * t + ln;
                aout[(size_t)row * D_MODEL + col] = (_Float16)(acc[qt][t][r] * inv);
            }
        }
    }
}

// ---------------- kernel 3: FC + bias (32x64 per wave), f32 out ----------------
__global__ __launch_bounds__(256) void MDIA_fc_gemm(
        const _Float16* __restrict__ ah, const _Float16* __restrict__ wfh,
        const float* __restrict__ bias, float* __restrict__ out) {
    const int lane = threadIdx.x & 31, wv = threadIdx.x >> 5;
    const int hi = lane >> 4, ln = lane & 15;
    const int m0 = blockIdx.x * 32;
    const int n0 = (blockIdx.y * 8 + wv) * 64;

    v8f acc[2][4] = {};
    for (int k0 = 0; k0 < D_MODEL; k0 += 32) {
        v16h a0 = load_a_frag(ah, D_MODEL, m0 + ln, k0, hi);
        v16h a1 = load_a_frag(ah, D_MODEL, m0 + 16 + ln, k0, hi);
#pragma unroll
        for (int nt = 0; nt < 4; ++nt) {
            v16h b = load_b_frag(wfh, D_MODEL, n0 + 16 * nt + ln, k0, hi);
            acc[0][nt] = wmma_f16(a0, b, acc[0][nt]);
            acc[1][nt] = wmma_f16(a1, b, acc[1][nt]);
        }
    }
#pragma unroll
    for (int nt = 0; nt < 4; ++nt) {
        int col = n0 + 16 * nt + ln;
        float bv = bias[col];
#pragma unroll
        for (int mt = 0; mt < 2; ++mt) {
#pragma unroll
            for (int r = 0; r < 8; ++r) {
                int row = m0 + 16 * mt + r + 8 * hi;
                out[(size_t)row * D_MODEL + col] = acc[mt][nt][r] + bv;
            }
        }
    }
}

extern "C" void kernel_launch(void* const* d_in, const int* in_sizes, int n_in,
                              void* d_out, int out_size, void* d_ws, size_t ws_size,
                              hipStream_t stream) {
    const float* x     = (const float*)d_in[0];   // [4,2048,1024]
    const float* w_qkv = (const float*)d_in[1];   // [3072,1024]
    const float* w_fc  = (const float*)d_in[2];   // [1024,1024]
    const float* b_fc  = (const float*)d_in[3];   // [1024]
    float* out = (float*)d_out;

    char* ws = (char*)d_ws;
    size_t off = 0;
    auto carve = [&](size_t bytes) -> char* {
        char* p = ws + off;
        off += (bytes + 255) & ~(size_t)255;
        return p;
    };
    _Float16* xh   = (_Float16*)carve((size_t)ROWS * D_MODEL * 2);       // 16 MB
    _Float16* wqh  = (_Float16*)carve((size_t)3 * D_MODEL * D_MODEL * 2);// 6 MB
    _Float16* wfh  = (_Float16*)carve((size_t)D_MODEL * D_MODEL * 2);    // 2 MB
    _Float16* qb   = (_Float16*)carve((size_t)ROWS * D_MODEL * 2);       // 16 MB
    _Float16* kb   = (_Float16*)carve((size_t)ROWS * D_MODEL * 2);       // 16 MB
    _Float16* vtb  = (_Float16*)carve((size_t)ROWS * D_MODEL * 2);       // 16 MB
    _Float16* aout = (_Float16*)carve((size_t)ROWS * D_MODEL * 2);       // 16 MB

    const int nx = ROWS * D_MODEL, nq = 3 * D_MODEL * D_MODEL, nf = D_MODEL * D_MODEL;
    MDIA_cvt_f16<<<(nx + 255) / 256, 256, 0, stream>>>(x, xh, nx);
    MDIA_cvt_f16<<<(nq + 255) / 256, 256, 0, stream>>>(w_qkv, wqh, nq);
    MDIA_cvt_f16<<<(nf + 255) / 256, 256, 0, stream>>>(w_fc, wfh, nf);

    MDIA_qkv_gemm<<<dim3(ROWS / 32, (3 * D_MODEL) / 512), 256, 0, stream>>>(
        xh, wqh, qb, kb, vtb);

    MDIA_attn<<<dim3(SEQ / 128, BATCH * NHEAD), 128, 0, stream>>>(qb, kb, vtb, aout);

    MDIA_fc_gemm<<<dim3(ROWS / 32, D_MODEL / 512), 256, 0, stream>>>(
        aout, wfh, b_fc, out);
}
